// Gate_77567109366011
// MI455X (gfx1250) — compile-verified, matrix operand order checked by reading, not run on hardware
//
#include <hip/hip_runtime.h>
#include <hip/hip_bf16.h>

typedef __attribute__((ext_vector_type(16))) __bf16 v16bf;
typedef __attribute__((ext_vector_type(8)))  float  v8f;

#define CC 256  // out channels (fixed by problem)

// ---------------------------------------------------------------- fill zero
__global__ void fill_zero_kernel(float* __restrict__ p, long n) {
    long i = (long)blockIdx.x * blockDim.x + threadIdx.x;
    long stride = (long)gridDim.x * blockDim.x;
    for (; i < n; i += stride) p[i] = 0.0f;
}

// ------------------------------------------------- edge mean-aggregation
// One wave per edge: gather 256 f32 of x[src] (2x float4 per lane, coalesced)
// and scatter-add into agg[dst] with f32 atomics (land in L2; the 102MB agg
// buffer is L2-resident in the 192MB global L2). Lane 0 bumps the counter.
__global__ __launch_bounds__(256) void agg_kernel(
    const float* __restrict__ x, const int* __restrict__ eidx,
    float* __restrict__ agg, float* __restrict__ cnt, int E) {
    int gtid = blockIdx.x * blockDim.x + threadIdx.x;
    int wave = gtid >> 5;
    int lane = gtid & 31;
    int nwaves = (gridDim.x * blockDim.x) >> 5;
    for (int e = wave; e < E; e += nwaves) {
        int src = eidx[e];
        int dst = eidx[E + e];
        const float4* xs = (const float4*)(x + (size_t)src * CC);
        float* ag = agg + (size_t)dst * CC;
        float4 v0 = xs[lane];
        float4 v1 = xs[lane + 32];
        int c0 = lane * 4;
        atomicAdd(&ag[c0 + 0], v0.x);
        atomicAdd(&ag[c0 + 1], v0.y);
        atomicAdd(&ag[c0 + 2], v0.z);
        atomicAdd(&ag[c0 + 3], v0.w);
        int c1 = 128 + lane * 4;
        atomicAdd(&ag[c1 + 0], v1.x);
        atomicAdd(&ag[c1 + 1], v1.y);
        atomicAdd(&ag[c1 + 2], v1.z);
        atomicAdd(&ag[c1 + 3], v1.w);
        if (lane == 0) atomicAdd(&cnt[dst], 1.0f);
    }
}

// --------------------------------------------------------- mean normalize
__global__ void normalize_kernel(float* __restrict__ agg,
                                 const float* __restrict__ cnt, long Nn) {
    long total = Nn * CC;
    long i = (long)blockIdx.x * blockDim.x + threadIdx.x;
    long stride = (long)gridDim.x * blockDim.x;
    for (; i < total; i += stride) {
        float c = cnt[i >> 8];  // i / 256
        agg[i] = agg[i] / fmaxf(c, 1.0f);
    }
}

// pack two f32 into adjacent bf16 halves (lowers to v_cvt_pk_bf16_f32)
__device__ __forceinline__ unsigned pk_bf16(float lo, float hi) {
    union { __bf16 h[2]; unsigned u; } p;
    p.h[0] = (__bf16)lo;
    p.h[1] = (__bf16)hi;
    return p.u;
}

// fast sigmoid: v_mul + v_exp + v_add + v_rcp (no IEEE div sequence)
__device__ __forceinline__ float fast_sigmoid(float v) {
    return __builtin_amdgcn_rcpf(1.0f + __expf(-v));
}

// ----------------------------------------------------------- WMMA GEMM
// Out[M x 256] = A1[M x K1] @ W1[K1 x 256] (+ A2[M x K2] @ W2[K2 x 256])
//                (+ bias1[256]) (+ bias2[256]) ; optional sigmoid.
// f32 operands packed to bf16 while staging to LDS;
// v_wmma_f32_16x16x32_bf16 with f32 accumulate.
// Block = 256 thr (8 waves) computes a full 32-row x 256-col strip so Out may
// alias A2 (in-place SAGE update): every K read of the strip completes
// (barrier) before the strip is written, and no other block touches it.
// Wave w: row-group w>>2 (16 rows), col-group w&3 (64 cols = 4 WMMA N-tiles).
__global__ __launch_bounds__(256) void gemm_wmma_kernel(
    const float* __restrict__ A1, const float* __restrict__ W1, int K1,
    const float* __restrict__ A2, const float* __restrict__ W2, int K2,
    const float* __restrict__ bias1, const float* __restrict__ bias2,
    float* __restrict__ Out, int M, int applySigmoid) {
    __shared__ __bf16 sA[32][36];    // [m][k], padded; k-contiguous
    __shared__ __bf16 sB[CC][36];    // [n][k], padded; (k,k+1) pairs contiguous

    const int row0 = blockIdx.x * 32;
    const int tid  = threadIdx.x;
    const int wv   = tid >> 5;
    const int lane = tid & 31;
    const int half = lane >> 4;      // 0: lanes 0-15, 1: lanes 16-31
    const int mn   = lane & 15;
    const int rg   = wv >> 2;        // row group 0/1
    const int cg   = wv & 3;         // col group 0..3

    // preload per-lane bias sums (uniform branches, 4 loads each, overlaps
    // the K-loop; keeps the epilogue branch-free)
    float badd[4] = {0.f, 0.f, 0.f, 0.f};
    if (bias1) {
#pragma unroll
        for (int t = 0; t < 4; t++) badd[t] += bias1[cg * 64 + t * 16 + mn];
    }
    if (bias2) {
#pragma unroll
        for (int t = 0; t < 4; t++) badd[t] += bias2[cg * 64 + t * 16 + mn];
    }

    v8f acc[4];
    v8f zero = {0.f, 0.f, 0.f, 0.f, 0.f, 0.f, 0.f, 0.f};
#pragma unroll
    for (int t = 0; t < 4; t++) acc[t] = zero;

#pragma unroll 1
    for (int phase = 0; phase < 2; phase++) {
        const float* A = phase ? A2 : A1;
        const float* W = phase ? W2 : W1;
        const int    K = phase ? K2 : K1;
        if (A == nullptr) continue;
        for (int k0 = 0; k0 < K; k0 += 32) {
            __syncthreads();
            // stage A tile 32x32: one float4 per thread, branchless row clamp,
            // packed bf16 pairs -> one 8B LDS store
            {
                int i = tid * 4;            // 0..1020
                int m = i >> 5, k = i & 31;
                int r = row0 + m;
                r = (r < M) ? r : (M - 1);  // clamp: no exec-mask branching
                float4 v = *(const float4*)&A[(size_t)r * K + (k0 + k)];
                uint2 pk;
                pk.x = pk_bf16(v.x, v.y);
                pk.y = pk_bf16(v.z, v.w);
                *(uint2*)&sA[m][k] = pk;
            }
            // stage W tile 32x256 into [n][k] layout: each step loads float4
            // from rows k=2*kp and k=2*kp+1 (coalesced over n), stores packed
            // (k,k+1) bf16 pairs as b32 exactly where the frag gather reads
#pragma unroll
            for (int it = 0; it < 4; it++) {
                int idx = it * 256 + tid;   // 0..1023
                int kp  = idx >> 6;         // k-pair 0..15
                int n   = (idx & 63) * 4;   // n quad
                const float* Wp = W + (size_t)(k0 + 2 * kp) * CC + n;
                float4 w0 = *(const float4*)Wp;
                float4 w1 = *(const float4*)(Wp + CC);
                *(unsigned*)&sB[n + 0][2 * kp] = pk_bf16(w0.x, w1.x);
                *(unsigned*)&sB[n + 1][2 * kp] = pk_bf16(w0.y, w1.y);
                *(unsigned*)&sB[n + 2][2 * kp] = pk_bf16(w0.z, w1.z);
                *(unsigned*)&sB[n + 3][2 * kp] = pk_bf16(w0.w, w1.w);
            }
            __syncthreads();
            // A fragment per ISA 16-bit 16x32 layout
            v16bf af;
#pragma unroll
            for (int e = 0; e < 16; e++) {
                int v = e >> 1, p = e & 1;
                int kk = (v < 4 ? 2 * v + p : 16 + 2 * (v - 4) + p) + 8 * half;
                af[e] = sA[rg * 16 + mn][kk];
            }
#pragma unroll
            for (int t = 0; t < 4; t++) {
                int col = cg * 64 + t * 16 + mn;
                v16bf bfr;
#pragma unroll
                for (int e = 0; e < 16; e++) {
                    int v = e >> 1, p = e & 1;
                    int kk = (v < 4 ? 2 * v + p : 16 + 2 * (v - 4) + p) + 8 * half;
                    bfr[e] = sB[col][kk];
                }
                acc[t] = __builtin_amdgcn_wmma_f32_16x16x32_bf16(
                    false, af, false, bfr, (short)0, acc[t], false, false);
            }
        }
    }
    // all reads of (possibly aliased) A2 strip are done before stores
    __syncthreads();
    const int mbase = row0 + rg * 16 + 8 * half;  // C/D: vgpr r -> M=r+8*half
    if (row0 + 32 <= M) {
        // fast path: full strip, uniform sigmoid branch hoisted out
        if (applySigmoid) {
#pragma unroll
            for (int t = 0; t < 4; t++) {
                int n = cg * 64 + t * 16 + mn;
#pragma unroll
                for (int r = 0; r < 8; r++) {
                    Out[(size_t)(mbase + r) * CC + n] =
                        fast_sigmoid(acc[t][r] + badd[t]);
                }
            }
        } else {
#pragma unroll
            for (int t = 0; t < 4; t++) {
                int n = cg * 64 + t * 16 + mn;
#pragma unroll
                for (int r = 0; r < 8; r++) {
                    Out[(size_t)(mbase + r) * CC + n] = acc[t][r] + badd[t];
                }
            }
        }
    } else {
#pragma unroll
        for (int t = 0; t < 4; t++) {
            int n = cg * 64 + t * 16 + mn;
#pragma unroll
            for (int r = 0; r < 8; r++) {
                int m = mbase + r;
                if (m < M) {
                    float v = acc[t][r] + badd[t];
                    if (applySigmoid) v = fast_sigmoid(v);
                    Out[(size_t)m * CC + n] = v;
                }
            }
        }
    }
}

extern "C" void kernel_launch(void* const* d_in, const int* in_sizes, int n_in,
                              void* d_out, int out_size, void* d_ws, size_t ws_size,
                              hipStream_t stream) {
    const float* x_A    = (const float*)d_in[0];
    const float* x_B    = (const float*)d_in[1];
    // d_in[2], d_in[3] = h_A, h_B : unused by the reference
    const int*   e_ab   = (const int*)d_in[4];
    const int*   e_ba   = (const int*)d_in[5];
    const float* lin_A  = (const float*)d_in[6];
    const float* lin_B  = (const float*)d_in[7];
    const float* bias_A = (const float*)d_in[8];
    const float* bias_B = (const float*)d_in[9];
    const float* Wl_ab0 = (const float*)d_in[10];
    const float* bl_ab0 = (const float*)d_in[11];
    const float* Wr_ab0 = (const float*)d_in[12];
    const float* Wl_ba0 = (const float*)d_in[13];
    const float* bl_ba0 = (const float*)d_in[14];
    const float* Wr_ba0 = (const float*)d_in[15];
    const float* Wl_ab1 = (const float*)d_in[16];
    const float* bl_ab1 = (const float*)d_in[17];
    const float* Wr_ab1 = (const float*)d_in[18];
    const float* Wl_ba1 = (const float*)d_in[19];
    const float* bl_ba1 = (const float*)d_in[20];
    const float* Wr_ba1 = (const float*)d_in[21];

    const int F = 128;
    const int N = in_sizes[0] / F;     // 100000
    const int E = in_sizes[4] / 2;     // 1600000
    const size_t NC = (size_t)N * CC;

    float* curA = (float*)d_ws;
    float* curB = curA + NC;
    float* aggA = curB + NC;
    float* aggB = aggA + NC;
    float* cntA = aggB + NC;
    float* cntB = cntA + N;
    float* outA = (float*)d_out;       // [N, C] gA
    float* outB = outA + NC;           // [N, C] gB

    const int gemmGrid = (N + 31) / 32;
    dim3 gblk(256);

    // input projection: cur = x @ lin  (K = 128)
    gemm_wmma_kernel<<<gemmGrid, gblk, 0, stream>>>(
        x_A, lin_A, F, nullptr, nullptr, 0, nullptr, nullptr, curA, N, 0);
    gemm_wmma_kernel<<<gemmGrid, gblk, 0, stream>>>(
        x_B, lin_B, F, nullptr, nullptr, 0, nullptr, nullptr, curB, N, 0);

    for (int l = 0; l < 2; l++) {
        // zero agg buffers + counters (contiguous region)
        fill_zero_kernel<<<2048, 256, 0, stream>>>(aggA, (long)(2 * NC + 2 * (size_t)N));
        // mean aggregation: A <- B via eidx_ba ; B <- A via eidx_ab
        agg_kernel<<<4096, 256, 0, stream>>>(curB, e_ba, aggA, cntA, E);
        agg_kernel<<<4096, 256, 0, stream>>>(curA, e_ab, aggB, cntB, E);
        normalize_kernel<<<4096, 256, 0, stream>>>(aggA, cntA, (long)N);
        normalize_kernel<<<4096, 256, 0, stream>>>(aggB, cntB, (long)N);
        if (l == 0) {
            // in-place SAGE update (cur strip fully consumed before stores)
            gemm_wmma_kernel<<<gemmGrid, gblk, 0, stream>>>(
                aggA, Wl_ba0, CC, curA, Wr_ba0, CC, bl_ba0, nullptr, curA, N, 0);
            gemm_wmma_kernel<<<gemmGrid, gblk, 0, stream>>>(
                aggB, Wl_ab0, CC, curB, Wr_ab0, CC, bl_ab0, nullptr, curB, N, 0);
        } else {
            // final layer fuses node-type bias + sigmoid, writes d_out
            gemm_wmma_kernel<<<gemmGrid, gblk, 0, stream>>>(
                aggA, Wl_ba1, CC, curA, Wr_ba1, CC, bl_ba1, bias_A, outA, N, 1);
            gemm_wmma_kernel<<<gemmGrid, gblk, 0, stream>>>(
                aggB, Wl_ab1, CC, curB, Wr_ab1, CC, bl_ab1, bias_B, outB, N, 1);
        }
    }
}
